// MultiHeadAttention_24876450578725
// MI455X (gfx1250) — matrix operand order repaired and checked by reading, not compiled, not run on hardware
//
#include <hip/hip_runtime.h>
#include <hip/hip_bf16.h>
#include <math.h>

typedef __attribute__((ext_vector_type(16))) _Float16 v16h;
typedef __attribute__((ext_vector_type(8)))  float    v8f;

#define D_MODEL 768
#define NHEAD   12
#define DK      64
#define SEQ     2048
#define BATCH   2
#define MTOT    (BATCH*SEQ)   /* 4096 */

// LDS generic pointer -> 32-bit LDS byte offset (aperture form: addr[31:0] is
// the LDS offset per CDNA5 ISA §10.2 aperture address calculation)
__device__ __forceinline__ unsigned lds_u32(const void* p) {
  return (unsigned)(uintptr_t)p;
}

// ---------------------------------------------------------------------------
// Y = X @ W^T + b    X:[M,768] (f32 or f16), W:[768,768] f32, b:[768] f32
// mode 0: f16 out, split-heads  [B,H,S,dk]      (Q and K)
// mode 2: f16 out, transposed   [B,H,dk,S]      (V)
// mode 3: f32 out, row-major    [M,768]         (final projection)
// f32->f16 conversion fused into the LDS fill, so staging stays in VGPR path.
// ---------------------------------------------------------------------------
template<typename TA>
__global__ __launch_bounds__(256) void linear_wmma(
    const TA* __restrict__ X, const float* __restrict__ W,
    const float* __restrict__ bias, void* __restrict__ Yout, int mode)
{
  constexpr int Kd = D_MODEL;
  __shared__ __align__(16) _Float16 lA[128][48];   // 128x32 tile, padded
  __shared__ __align__(16) _Float16 lB[64][48];    // 64x32 tile (rows = n)

  const int tid = threadIdx.x;
  const int l   = tid & 31, wid = tid >> 5;
  const int r   = l & 15,  hi  = l >> 4;
  const int wm  = wid & 3, wn  = wid >> 2;      // 4x2 wave grid
  const int m0  = blockIdx.x * 128;
  const int n0  = blockIdx.y * 64;

  v8f acc[2][2] = {};

  const int rowA = tid >> 1, segA = (tid & 1) * 16;   // 16 elems / thread
  const int rowB = tid >> 2, segB = (tid & 3) * 8;    // 8 elems / thread

  for (int k0 = 0; k0 < Kd; k0 += 32) {
    __syncthreads();
    {   // stage A tile (convert to f16 on the fly)
      const TA* src = X + (size_t)(m0 + rowA) * Kd + k0 + segA;
      _Float16* dst = &lA[rowA][segA];
      #pragma unroll
      for (int i = 0; i < 16; ++i) dst[i] = (_Float16)src[i];
    }
    {   // stage B tile from W (row n, contiguous k) with f32->f16 convert
      const float* src = W + (size_t)(n0 + rowB) * Kd + k0 + segB;
      _Float16* dst = &lB[rowB][segB];
      #pragma unroll
      for (int i = 0; i < 8; ++i) dst[i] = (_Float16)src[i];
    }
    __syncthreads();

    v16h af[2], bf[2];
    #pragma unroll
    for (int mt = 0; mt < 2; ++mt) {            // A frag: 16 contiguous k per lane
      const _Float16* pa = &lA[wm*32 + mt*16 + r][hi*16];
      #pragma unroll
      for (int j = 0; j < 8; ++j) { af[mt][j] = pa[j]; af[mt][j+8] = pa[8+j]; }
    }
    #pragma unroll
    for (int nt = 0; nt < 2; ++nt) {            // B frag: 16 contiguous k per lane
      const _Float16* pb = &lB[wn*32 + nt*16 + r][hi*16];
      #pragma unroll
      for (int j = 0; j < 16; ++j) bf[nt][j] = pb[j];
    }
    #pragma unroll
    for (int mt = 0; mt < 2; ++mt)
      #pragma unroll
      for (int nt = 0; nt < 2; ++nt)
        acc[mt][nt] = __builtin_amdgcn_wmma_f32_16x16x32_f16(
            false, af[mt], false, bf[nt], (short)0, acc[mt][nt], false, false);
  }

  // epilogue: bias + layout scatter (C/D layout: lane holds col r, rows hi*8+i)
  #pragma unroll
  for (int mt = 0; mt < 2; ++mt) {
    #pragma unroll
    for (int nt = 0; nt < 2; ++nt) {
      const int n = n0 + wn*32 + nt*16 + r;
      const float bn = bias[n];
      const int h = n >> 6, d = n & 63;
      #pragma unroll
      for (int i = 0; i < 8; ++i) {
        const int m = m0 + wm*32 + mt*16 + hi*8 + i;
        const int b = m >> 11, s = m & (SEQ - 1);
        const float v = acc[mt][nt][i] + bn;
        if (mode == 3) {
          ((float*)Yout)[(size_t)m * D_MODEL + n] = v;
        } else if (mode == 0) {
          ((_Float16*)Yout)[(((size_t)b*NHEAD + h)*SEQ + s)*DK + d] = (_Float16)v;
        } else { // mode 2: V transposed per head
          ((_Float16*)Yout)[(((size_t)b*NHEAD + h)*DK + d)*SEQ + s] = (_Float16)v;
        }
      }
    }
  }
}

// ---------------------------------------------------------------------------
// Flash-attention style: per (b,h), 128 query rows per block (16 per wave),
// online softmax over 64-key blocks. K/V blocks staged memory->LDS with the
// CDNA5 async engine (GLOBAL_LOAD_ASYNC_TO_LDS_B128, ASYNCcnt) — no VGPR
// round-trip. ctx written transposed [bh][dk][S] so the reference's
// transpose(0,1,3,2).reshape(B,S,768) is an identity reinterpret.
// Q:[bh,S,dk]  K:[bh,S,dk]  Vt:[bh,dk,S]  Ctx:[bh,dk,S]   all f16
// ---------------------------------------------------------------------------
__global__ __launch_bounds__(256) void attn_wmma(
    const _Float16* __restrict__ Q, const _Float16* __restrict__ Kh,
    const _Float16* __restrict__ Vt, _Float16* __restrict__ Ctx)
{
  __shared__ __align__(16) _Float16 lK[64][72];       // [key][d]
  __shared__ __align__(16) _Float16 lV[64][72];       // [d][key]
  __shared__ __align__(16) _Float16 lP[8][16][72];    // per-wave P staging

  const int tid = threadIdx.x;
  const int l   = tid & 31, wid = tid >> 5;
  const int r   = l & 15,  hi  = l >> 4;
  const int bh  = blockIdx.y;
  const int qbase = blockIdx.x * 128 + wid * 16;

  const _Float16 qscale = (_Float16)0.125f;           // 1/sqrt(64)

  // preload Q fragments (two 16x32 d-chunks), fold in softmax scale
  v16h qa[2];
  {
    const _Float16* pq = Q + ((size_t)bh * SEQ + qbase + r) * DK;
    #pragma unroll
    for (int dc = 0; dc < 2; ++dc) {
      #pragma unroll
      for (int j = 0; j < 8; ++j) {
        qa[dc][j]   = pq[dc*32 + hi*16 + j]      * qscale;
        qa[dc][j+8] = pq[dc*32 + hi*16 + 8 + j] * qscale;
      }
    }
  }

  v8f cacc[4] = {};
  float mrun[8], srun[8];
  #pragma unroll
  for (int i = 0; i < 8; ++i) { mrun[i] = -1e30f; srun[i] = 0.f; }

  const int rowL = tid >> 2, segL = (tid & 3) * 16;   // 64 rows x 4 segs of 16

  // per-thread LDS destinations and uniform global bases for async staging
  const unsigned dstK = lds_u32(&lK[rowL][segL]);
  const unsigned dstV = lds_u32(&lV[rowL][segL]);
  const unsigned long long baseK = (unsigned long long)(uintptr_t)Kh;
  const unsigned long long baseV = (unsigned long long)(uintptr_t)Vt;
  // byte offsets; the kb-dependent part is added per iteration
  const unsigned offK0 = (unsigned)((((size_t)bh*SEQ + rowL) * DK + segL) * sizeof(_Float16));
  const unsigned offV0 = (unsigned)((((size_t)bh*DK  + rowL) * SEQ + segL) * sizeof(_Float16));

  for (int kb = 0; kb < SEQ; kb += 64) {
    __syncthreads();   // previous block fully consumed before overwrite
    {
      const unsigned offK = offK0 + (unsigned)(kb * DK * sizeof(_Float16)); // keys advance rows
      const unsigned offV = offV0 + (unsigned)(kb * sizeof(_Float16));      // keys advance cols
      // 32 B per lane per matrix: two B128 async copies, INST_OFFSET applies
      // to both LDS and global addresses (ISA §10.7 async pseudocode).
      asm volatile(
        "global_load_async_to_lds_b128 %0, %2, %4\n\t"
        "global_load_async_to_lds_b128 %0, %2, %4 offset:16\n\t"
        "global_load_async_to_lds_b128 %1, %3, %5\n\t"
        "global_load_async_to_lds_b128 %1, %3, %5 offset:16\n\t"
        "s_wait_asynccnt 0x0"
        :
        : "v"(dstK), "v"(dstV), "v"(offK), "v"(offV), "s"(baseK), "s"(baseV)
        : "memory");
    }
    __syncthreads();   // LDS tiles visible to all waves

    // scores: 16 q x 64 keys = 4 C tiles, 2 d-chunk WMMAs each
    v8f sc[4] = {};
    #pragma unroll
    for (int t = 0; t < 4; ++t) {
      #pragma unroll
      for (int dc = 0; dc < 2; ++dc) {
        v16h bf;
        const _Float16* pb = &lK[t*16 + r][dc*32 + hi*16];
        #pragma unroll
        for (int j = 0; j < 16; ++j) bf[j] = pb[j];
        sc[t] = __builtin_amdgcn_wmma_f32_16x16x32_f16(
            false, qa[dc], false, bf, (short)0, sc[t], false, false);
      }
    }

    // online softmax (rows live across 16 lanes -> xor-reduce 1/2/4/8)
    float scale8[8];
    #pragma unroll
    for (int i = 0; i < 8; ++i) {
      float v = fmaxf(fmaxf(sc[0][i], sc[1][i]), fmaxf(sc[2][i], sc[3][i]));
      v = fmaxf(v, __shfl_xor(v, 1, 32));
      v = fmaxf(v, __shfl_xor(v, 2, 32));
      v = fmaxf(v, __shfl_xor(v, 4, 32));
      v = fmaxf(v, __shfl_xor(v, 8, 32));
      const float mnew = fmaxf(mrun[i], v);
      scale8[i] = __expf(mrun[i] - mnew);
      mrun[i] = mnew;
      float ps = 0.f;
      #pragma unroll
      for (int t = 0; t < 4; ++t) {
        const float p = __expf(sc[t][i] - mnew);
        ps += p;
        lP[wid][hi*8 + i][t*16 + r] = (_Float16)p;   // C-layout -> LDS
      }
      ps += __shfl_xor(ps, 1, 32);
      ps += __shfl_xor(ps, 2, 32);
      ps += __shfl_xor(ps, 4, 32);
      ps += __shfl_xor(ps, 8, 32);
      srun[i] = srun[i] * scale8[i] + ps;
    }
    #pragma unroll
    for (int dt = 0; dt < 4; ++dt)
      #pragma unroll
      for (int i = 0; i < 8; ++i) cacc[dt][i] *= scale8[i];

    // ctx += P @ V  (re-read P from LDS in A-fragment layout; per-wave region,
    // same-wave LDS ops are in-order so no barrier needed)
    #pragma unroll
    for (int kc = 0; kc < 2; ++kc) {
      v16h pf;
      const _Float16* pp = &lP[wid][r][kc*32];
      #pragma unroll
      for (int j = 0; j < 8; ++j) { pf[j] = pp[hi*16 + j]; pf[j+8] = pp[hi*16 + 8 + j]; }
      #pragma unroll
      for (int dt = 0; dt < 4; ++dt) {
        v16h bf;
        const _Float16* pb = &lV[dt*16 + r][kc*32 + hi*16];
        #pragma unroll
        for (int j = 0; j < 16; ++j) bf[j] = pb[j];
        cacc[dt] = __builtin_amdgcn_wmma_f32_16x16x32_f16(
            false, pf, false, bf, (short)0, cacc[dt], false, false);
      }
    }
  }

  // normalize and store transposed ctx [bh][d][q]
  float inv[8];
  #pragma unroll
  for (int i = 0; i < 8; ++i) inv[i] = 1.f / srun[i];
  #pragma unroll
  for (int dt = 0; dt < 4; ++dt) {
    const int d = dt*16 + r;
    #pragma unroll
    for (int i = 0; i < 8; ++i) {
      const int q = qbase + hi*8 + i;
      Ctx[((size_t)bh*DK + d)*SEQ + q] = (_Float16)(cacc[dt][i] * inv[i]);
    }
  }
}

// ---------------------------------------------------------------------------
extern "C" void kernel_launch(void* const* d_in, const int* in_sizes, int n_in,
                              void* d_out, int out_size, void* d_ws, size_t ws_size,
                              hipStream_t stream) {
  const float* q  = (const float*)d_in[0];
  const float* k  = (const float*)d_in[1];
  const float* v  = (const float*)d_in[2];
  const float* Wq = (const float*)d_in[3];
  const float* bq = (const float*)d_in[4];
  const float* Wk = (const float*)d_in[5];
  const float* bk = (const float*)d_in[6];
  const float* Wv = (const float*)d_in[7];
  const float* bv = (const float*)d_in[8];
  const float* Wo = (const float*)d_in[9];
  const float* bo = (const float*)d_in[10];

  const size_t NELT = (size_t)MTOT * D_MODEL;          // 3,145,728
  _Float16* Q16 = (_Float16*)d_ws;
  _Float16* K16 = Q16 + NELT;
  _Float16* V16 = K16 + NELT;
  _Float16* C16 = V16 + NELT;                          // ~25 MB total

  dim3 gg(MTOT/128, D_MODEL/64);                       // 32 x 12
  linear_wmma<float><<<gg, 256, 0, stream>>>(q, Wq, bq, (void*)Q16, 0);
  linear_wmma<float><<<gg, 256, 0, stream>>>(k, Wk, bk, (void*)K16, 0);
  linear_wmma<float><<<gg, 256, 0, stream>>>(v, Wv, bv, (void*)V16, 2);

  dim3 ga(SEQ/128, BATCH*NHEAD);                       // 16 x 24
  attn_wmma<<<ga, 256, 0, stream>>>(Q16, K16, V16, C16);

  linear_wmma<_Float16><<<gg, 256, 0, stream>>>(C16, Wo, bo, d_out, 3);
}
